// SwitchHead_21715354649836
// MI455X (gfx1250) — matrix operand order repaired and checked by reading, not compiled, not run on hardware
//
#include <hip/hip_runtime.h>

// ---------------------------------------------------------------------------
// SwitchHead fused pipeline for MI455X (gfx1250, wave32, BF16 WMMA).
//   K0: weights f32 -> bf16 (once, 590KB)
//   K1: keys  = MLP(switches)            [98304,256]->relu256->[.,128]
//   K2: query = MLP(auto_emb)            [16384,256]->relu256->[.,128]
//   K3: logits/softmax/sample/gather     (tiny; gathers bf16)
//   K4: proj  = MLP(gathered) + residual [16384,128]->relu256->[.,256]
// GEMMs: v_wmma_f32_16x16x32_bf16. A-tiles staged once per block into LDS
// (bf16), B read directly as pre-converted bf16 -> no cvt in inner loops.
// ---------------------------------------------------------------------------

typedef __attribute__((ext_vector_type(16))) __bf16          v16bf;
typedef __attribute__((ext_vector_type(8)))  float           v8f;
typedef __attribute__((ext_vector_type(8)))  unsigned short  ushort8v;

#define NW 8  // waves per 256-thread block

union BF16x16 { unsigned short u[16]; v16bf v; };

#define LD4(p) (*(const float4*)(p))

// pack two floats into two bf16 (backend should use hw cvt, e.g. v_cvt_pk_bf16_f32)
__device__ __forceinline__ unsigned pack2_bf16(float a, float b) {
  union { __bf16 h[2]; unsigned u; } r;
  r.h[0] = (__bf16)a; r.h[1] = (__bf16)b;
  return r.u;
}

__device__ __forceinline__ unsigned short to_bf16u(float a) {
  union { __bf16 h; unsigned short u; } r;
  r.h = (__bf16)a;
  return r.u;
}

// A-fragment from bf16 LDS: elements 0..7 = [b0..b0+7], 8..15 = [b1..b1+7]
__device__ __forceinline__ v16bf lds_a_frag(const unsigned short* s, int b0, int b1) {
  ushort8v h0 = *(const ushort8v*)(s + b0);
  ushort8v h1 = *(const ushort8v*)(s + b1);
  BF16x16 r;
#pragma unroll
  for (int i = 0; i < 8; ++i) { r.u[i] = h0[i]; r.u[8 + i] = h1[i]; }
  return r.v;
}

// B-fragment: 16 contiguous bf16 from pre-converted weights (two b128 loads)
__device__ __forceinline__ v16bf ldg_bf16_16(const unsigned short* p) {
  ushort8v h0 = *(const ushort8v*)p;
  ushort8v h1 = *(const ushort8v*)(p + 8);
  BF16x16 r;
#pragma unroll
  for (int i = 0; i < 8; ++i) { r.u[i] = h0[i]; r.u[8 + i] = h1[i]; }
  return r.v;
}

__device__ __forceinline__ v8f wmma_bf16(v16bf a, v16bf b, v8f c) {
  return __builtin_amdgcn_wmma_f32_16x16x32_bf16(false, a, false, b,
                                                 (short)0, c, false, false);
}

// ---------------------------------------------------------------------------
// weight conversion: f32 -> bf16, 4 elements / thread
// ---------------------------------------------------------------------------
__global__ __launch_bounds__(256)
void cvt_bf16_kernel(const float* __restrict__ src, unsigned short* __restrict__ dst, int n) {
  const int e = (blockIdx.x * 256 + threadIdx.x) * 4;
  if (e < n) {
    float4 f = LD4(src + e);
    uint2 p;
    p.x = pack2_bf16(f.x, f.y);
    p.y = pack2_bf16(f.z, f.w);
    *(uint2*)&dst[e] = p;
  }
}

// ---------------------------------------------------------------------------
// Two-layer MLP: Y[16-row tile] = relu(X @ W1^T + b1) @ W2^T + b2
//   SRC_BF16: X is bf16 (gathered embeddings); else f32 with on-the-fly cvt
//   FINAL:    Y = resid + (act==1) * (...)
// Block = 256 threads (8 waves) per 16-row M-tile. W1/W2 are bf16.
// ---------------------------------------------------------------------------
template<int IN, int HID, int OUT, bool SRC_BF16, bool FINAL>
__global__ __launch_bounds__(256)
void mlp2_kernel(const void* __restrict__ Xv,
                 const unsigned short* __restrict__ W1, const float* __restrict__ B1,
                 const unsigned short* __restrict__ W2, const float* __restrict__ B2,
                 float* __restrict__ Y,
                 const float* __restrict__ resid, const int* __restrict__ act) {
  constexpr int KT1 = IN / 32;
  constexpr int KT2 = HID / 32;
  constexpr int J1  = HID / 16 / NW;   // hidden N-tiles per wave
  constexpr int J2  = OUT / 16 / NW;   // output N-tiles per wave

  __shared__ unsigned short s_x[16 * IN];      // bf16 input tile
  __shared__ unsigned short s_hid[16 * HID];   // bf16 hidden activations

  const int tid  = threadIdx.x;
  const int wave = tid >> 5;
  const int lane = tid & 31;
  const int n    = lane & 15;     // column within tile / A-row within tile
  const int hs   = lane >> 4;     // half-wave selector
  const long row0 = (long)blockIdx.x * 16;
  const v8f vzero = {};

  // ---- stage 0: cooperative tile load -> bf16 LDS (once per block) ----
  if (SRC_BF16) {
    const unsigned short* X = (const unsigned short*)Xv;
    constexpr int CH = (16 * IN) / (8 * 256);          // uint4 = 8 bf16
#pragma unroll
    for (int c = 0; c < CH; ++c) {
      const int off = (tid + c * 256) * 8;
      const int row = off / IN, col = off % IN;
      *(uint4*)&s_x[row * IN + col] =
          *(const uint4*)(X + (size_t)(row0 + row) * IN + col);
    }
  } else {
    const float* X = (const float*)Xv;
    constexpr int CH = (16 * IN) / (4 * 256);          // float4 chunks
#pragma unroll
    for (int c = 0; c < CH; ++c) {
      const int off = (tid + c * 256) * 4;
      const int row = off / IN, col = off % IN;
      float4 f = LD4(X + (size_t)(row0 + row) * IN + col);
      uint2 p;
      p.x = pack2_bf16(f.x, f.y);
      p.y = pack2_bf16(f.z, f.w);
      *(uint2*)&s_x[row * IN + col] = p;
    }
  }
  __syncthreads();

  // ---- stage 1: hidden = relu(X @ W1^T + b1) -> LDS (bf16) ----
  v8f acc1[J1];
#pragma unroll
  for (int j = 0; j < J1; ++j) acc1[j] = vzero;

#pragma unroll
  for (int kt = 0; kt < KT1; ++kt) {
    const int akb = kt * 32 + hs * 16;
    v16bf a = lds_a_frag(s_x, n * IN + akb, n * IN + akb + 8);
    const int bkb = kt * 32 + hs * 16;
#pragma unroll
    for (int j = 0; j < J1; ++j) {
      const int nt = wave + j * NW;
      v16bf b = ldg_bf16_16(W1 + (size_t)(nt * 16 + n) * IN + bkb);
      acc1[j] = wmma_bf16(a, b, acc1[j]);
    }
  }

#pragma unroll
  for (int j = 0; j < J1; ++j) {
    const int nt = wave + j * NW;
    const float bias = B1[nt * 16 + n];
#pragma unroll
    for (int v = 0; v < 8; ++v) {
      const int mm = v + hs * 8;
      float h = acc1[j][v] + bias;
      h = h > 0.f ? h : 0.f;
      s_hid[mm * HID + nt * 16 + n] = to_bf16u(h);
    }
  }
  __syncthreads();

  // ---- stage 2: out = hidden @ W2^T + b2 (+ fused epilogue) ----
  v8f acc2[J2];
#pragma unroll
  for (int j = 0; j < J2; ++j) acc2[j] = vzero;

#pragma unroll
  for (int kt = 0; kt < KT2; ++kt) {
    const int akb = kt * 32 + hs * 16;
    v16bf a = lds_a_frag(s_hid, n * HID + akb, n * HID + akb + 8);
    const int bkb = kt * 32 + hs * 16;
#pragma unroll
    for (int j = 0; j < J2; ++j) {
      const int nt = wave + j * NW;
      v16bf b = ldg_bf16_16(W2 + (size_t)(nt * 16 + n) * HID + bkb);
      acc2[j] = wmma_bf16(a, b, acc2[j]);
    }
  }

#pragma unroll
  for (int j = 0; j < J2; ++j) {
    const int nt = wave + j * NW;
    const int o  = nt * 16 + n;
    const float bias = B2[o];
#pragma unroll
    for (int v = 0; v < 8; ++v) {
      const int mm = v + hs * 8;
      const long r = row0 + mm;
      float val = acc2[j][v] + bias;
      if (FINAL) {
        val = (act[r] == 1) ? val : 0.f;
        val += resid[(size_t)r * OUT + o];
      }
      Y[(size_t)r * OUT + o] = val;   // two contiguous 64B segments per VGPR
    }
  }
}

// ---------------------------------------------------------------------------
// Per-(t,b): 6 dot-128 logits, masked softmax, deterministic hash sample,
// gather chosen key row (as bf16). 1 block = 128 threads = one (t,b).
// ---------------------------------------------------------------------------
__global__ __launch_bounds__(128)
void attend_kernel(const float* __restrict__ Q, const float* __restrict__ K,
                   const int* __restrict__ mask,
                   float* __restrict__ o_logits, float* __restrict__ o_policy,
                   float* __restrict__ o_index, unsigned short* __restrict__ gathered) {
  const int tb   = blockIdx.x;
  const int t    = threadIdx.x;        // 0..127 == key-dim index
  const int wave = t >> 5, lane = t & 31;
  __shared__ float s_red[6][4];
  __shared__ int   s_idx;

  const float qv = Q[(size_t)tb * 128 + t];
  float p[6];
#pragma unroll
  for (int s = 0; s < 6; ++s)
    p[s] = qv * K[((size_t)tb * 6 + s) * 128 + t];
#pragma unroll
  for (int s = 0; s < 6; ++s) {
#pragma unroll
    for (int off = 16; off >= 1; off >>= 1)
      p[s] += __shfl_xor(p[s], off, 32);
  }
  if (lane == 0) {
#pragma unroll
    for (int s = 0; s < 6; ++s) s_red[s][wave] = p[s];
  }
  __syncthreads();

  if (t == 0) {
    float lg[6]; bool mk[6]; bool any = false;
#pragma unroll
    for (int s = 0; s < 6; ++s) {
      lg[s] = s_red[s][0] + s_red[s][1] + s_red[s][2] + s_red[s][3];
      mk[s] = mask[tb * 6 + s] != 0;
      any |= mk[s];
      o_logits[tb * 6 + s] = lg[s];            // raw logits output
    }
    if (!any) {
#pragma unroll
      for (int s = 0; s < 6; ++s) mk[s] = true;
    }
    float lmin = lg[0];
#pragma unroll
    for (int s = 1; s < 6; ++s) lmin = fminf(lmin, lg[s]);
    float m2[6]; float lmax = -3.4e38f;
#pragma unroll
    for (int s = 0; s < 6; ++s) { m2[s] = mk[s] ? lg[s] : lmin; lmax = fmaxf(lmax, m2[s]); }
    float ex[6]; float sum = 0.f;
#pragma unroll
    for (int s = 0; s < 6; ++s) { ex[s] = mk[s] ? expf(m2[s] - lmax) : 0.f; sum += ex[s]; }
    const float inv = 1.f / sum;
#pragma unroll
    for (int s = 0; s < 6; ++s) o_policy[tb * 6 + s] = ex[s] * inv;

    // deterministic PCG-style hash -> uniform in [0,1) -> inverse-CDF sample
    unsigned h = (unsigned)tb * 747796405u + 2891336453u;
    unsigned w = ((h >> ((h >> 28) + 4u)) ^ h) * 277803737u;
    w = (w >> 22) ^ w;
    const float u = (float)(w >> 8) * (1.0f / 16777216.0f);
    int idx = 5; float c = 0.f;
    for (int s = 0; s < 6; ++s) { c += ex[s] * inv; if (u < c) { idx = s; break; } }
    o_index[tb] = (float)idx;
    s_idx = idx;
  }
  __syncthreads();

  const int idx = s_idx;
  gathered[(size_t)tb * 128 + t] =
      to_bf16u(K[((size_t)tb * 6 + idx) * 128 + t]);
}

// ---------------------------------------------------------------------------
extern "C" void kernel_launch(void* const* d_in, const int* in_sizes, int n_in,
                              void* d_out, int out_size, void* d_ws, size_t ws_size,
                              hipStream_t stream) {
  const int*   act  = (const int*)  d_in[0];   // [T,B]
  const float* aemb = (const float*)d_in[1];   // [T,B,256]
  const float* sw   = (const float*)d_in[2];   // [T,B,S,256]
  const int*   mask = (const int*)  d_in[3];   // [T,B,S]
  const float* Wk1 = (const float*)d_in[4],  *bk1 = (const float*)d_in[5];
  const float* Wk2 = (const float*)d_in[6],  *bk2 = (const float*)d_in[7];
  const float* Wq1 = (const float*)d_in[8],  *bq1 = (const float*)d_in[9];
  const float* Wq2 = (const float*)d_in[10], *bq2 = (const float*)d_in[11];
  const float* Wp1 = (const float*)d_in[12], *bp1 = (const float*)d_in[13];
  const float* Wp2 = (const float*)d_in[14], *bp2 = (const float*)d_in[15];
  float* out = (float*)d_out;

  const int T = 64, B = 256, S = 6;
  const int TB = T * B;                 // 16384
  const int ROWS_K = TB * S;            // 98304

  // workspace layout:
  //   keys f32 [98304,128] | query f32 [16384,128] | gathered bf16 [16384,128]
  //   | bf16 weights (Wk1,Wk2,Wq1,Wq2,Wp1,Wp2)
  float* ws_k = (float*)d_ws;
  float* ws_q = ws_k + (size_t)ROWS_K * 128;
  unsigned short* ws_g = (unsigned short*)(ws_q + (size_t)TB * 128);
  unsigned short* wb   = ws_g + (size_t)TB * 128;
  unsigned short* bWk1 = wb;               wb += 256 * 256;
  unsigned short* bWk2 = wb;               wb += 128 * 256;
  unsigned short* bWq1 = wb;               wb += 256 * 256;
  unsigned short* bWq2 = wb;               wb += 128 * 256;
  unsigned short* bWp1 = wb;               wb += 256 * 128;
  unsigned short* bWp2 = wb;               wb += 256 * 256;

  float* o_logits = out;                                 // [T,B,S]
  float* o_policy = out + (size_t)TB * S;                // [T,B,S]
  float* o_index  = out + (size_t)TB * S * 2;            // [T,B,1]
  float* o_emb    = out + (size_t)TB * S * 2 + TB;       // [T,B,256]

  // K0: weight conversion (tiny)
  auto cvt = [&](const float* s, unsigned short* d, int n) {
    cvt_bf16_kernel<<<(n / 4 + 255) / 256, 256, 0, stream>>>(s, d, n);
  };
  cvt(Wk1, bWk1, 256 * 256);
  cvt(Wk2, bWk2, 128 * 256);
  cvt(Wq1, bWq1, 256 * 256);
  cvt(Wq2, bWq2, 128 * 256);
  cvt(Wp1, bWp1, 256 * 128);
  cvt(Wp2, bWp2, 256 * 256);

  mlp2_kernel<256, 256, 128, false, false><<<ROWS_K / 16, 256, 0, stream>>>(
      sw, bWk1, bk1, bWk2, bk2, ws_k, nullptr, nullptr);
  mlp2_kernel<256, 256, 128, false, false><<<TB / 16, 256, 0, stream>>>(
      aemb, bWq1, bq1, bWq2, bq2, ws_q, nullptr, nullptr);
  attend_kernel<<<TB, 128, 0, stream>>>(
      ws_q, ws_k, mask, o_logits, o_policy, o_index, ws_g);
  mlp2_kernel<128, 256, 256, true, true><<<TB / 16, 256, 0, stream>>>(
      ws_g, bWp1, bp1, bWp2, bp2, o_emb, aemb, act);
}
